// PointnetSAModuleMSGSSD_81037442941235
// MI455X (gfx1250) — matrix operand order repaired and checked by reading, not compiled, not run on hardware
//
#include <hip/hip_runtime.h>

#define BB 4
#define NN 16384
#define PP 2048
#define CINC 64
#define EPSB 1e-5f

typedef __attribute__((ext_vector_type(16))) _Float16 v16h;
typedef __attribute__((ext_vector_type(8)))  _Float16 v8h;
typedef __attribute__((ext_vector_type(8)))  float    v8f;
typedef __attribute__((ext_vector_type(4)))  unsigned int v4u;
typedef __attribute__((ext_vector_type(8)))  int      v8i;
typedef __attribute__((ext_vector_type(4)))  int      v4i;

union Frag16 { v16h v; v8h h[2]; };

// TDM builtin: 5-arg on ROCm 7.2 (clang-22), 6-arg on clang-23+ (therock HIP).
#if defined(__AMDGCN__) && __has_builtin(__builtin_amdgcn_tensor_load_to_lds)
#define USE_TDM 1
#if defined(__clang_major__) && (__clang_major__ >= 23)
#define TDM_ARGS6 1
#else
#define TDM_ARGS6 0
#endif
#else
#define USE_TDM 0
#define TDM_ARGS6 0
#endif

// ---------------------------------------------------------------------------
// 1. Farthest point sampling: one workgroup per batch, xyz resident in LDS
//    (192KB of the 320KB CDNA5 WGP LDS), per-thread dists in registers.
// ---------------------------------------------------------------------------
__global__ __launch_bounds__(1024)
void fps_kernel(const float* __restrict__ xyz, float* __restrict__ out) {
  extern __shared__ float sxyz[];            // sx[NN], sy[NN], sz[NN]
  float* sx = sxyz;
  float* sy = sxyz + NN;
  float* sz = sxyz + 2 * NN;
  __shared__ float swd[32];
  __shared__ int   swi[32];
  __shared__ int   sfar;

  const int b = blockIdx.x, tid = threadIdx.x;
  const float* base = xyz + (size_t)b * NN * 3;
  for (int i = tid; i < NN; i += 1024) {
    sx[i] = base[i * 3 + 0];
    sy[i] = base[i * 3 + 1];
    sz[i] = base[i * 3 + 2];
  }
  if (tid == 0) sfar = 0;

  float d[16];
#pragma unroll
  for (int j = 0; j < 16; ++j) d[j] = 1e10f;
  __syncthreads();

  float* oxyz = out;                               // (B,P,3)
  float* oind = out + BB * PP * 3 + BB * 128 * PP; // (B,P) as float of value

  for (int it = 0; it < PP; ++it) {
    const int far = sfar;
    const float cx = sx[far], cy = sy[far], cz = sz[far];
    if (tid == 0) {
      size_t o = ((size_t)b * PP + it);
      oxyz[o * 3 + 0] = cx;
      oxyz[o * 3 + 1] = cy;
      oxyz[o * 3 + 2] = cz;
      oind[o] = (float)far;
    }
    float best = -1.0f; int bi = 0;
#pragma unroll
    for (int j = 0; j < 16; ++j) {
      const int i = j * 1024 + tid;
      const float dx = sx[i] - cx, dy = sy[i] - cy, dz = sz[i] - cz;
      const float dd = dx * dx + dy * dy + dz * dz;
      d[j] = fminf(d[j], dd);
      if (d[j] > best) { best = d[j]; bi = i; }
    }
    // wave32 reduction
    for (int o = 16; o; o >>= 1) {
      float ob = __shfl_down(best, o, 32);
      int   oi = __shfl_down(bi,   o, 32);
      if (ob > best) { best = ob; bi = oi; }
    }
    const int wv = tid >> 5;
    if ((tid & 31) == 0) { swd[wv] = best; swi[wv] = bi; }
    __syncthreads();
    if (tid < 32) {
      best = swd[tid]; bi = swi[tid];
      for (int o = 16; o; o >>= 1) {
        float ob = __shfl_down(best, o, 32);
        int   oi = __shfl_down(bi,   o, 32);
        if (ob > best) { best = ob; bi = oi; }
      }
      if (tid == 0) sfar = bi;
    }
    __syncthreads();
  }
}

// ---------------------------------------------------------------------------
// 2. Ball query: one wave per query point, ballot_w32 prefix to collect the
//    first S neighbors in ascending index order (wave32-native).
// ---------------------------------------------------------------------------
__global__ void ball_query(const float* __restrict__ xyz,
                           const float* __restrict__ newxyz,
                           int* __restrict__ idxb, float r2, int S) {
  const int gw   = (int)((blockIdx.x * blockDim.x + threadIdx.x) >> 5);
  const int lane = threadIdx.x & 31;
  const int b = gw / PP;
  const float* nx = newxyz + (size_t)gw * 3;
  const float cx = nx[0], cy = nx[1], cz = nx[2];
  const float* px = xyz + (size_t)b * NN * 3;
  int* outp = idxb + (size_t)gw * S;

  int cnt = 0, first = -1;
  for (int basei = 0; basei < NN; basei += 32) {
    const int i = basei + lane;
    const float dx = px[i * 3 + 0] - cx;
    const float dy = px[i * 3 + 1] - cy;
    const float dz = px[i * 3 + 2] - cz;
    const bool hit = (dx * dx + dy * dy + dz * dz) <= r2;
    const unsigned mask = __builtin_amdgcn_ballot_w32(hit);
    if (first < 0 && mask) first = basei + __ffs(mask) - 1;
    const int before = cnt + __popc(mask & ((1u << lane) - 1u));
    if (hit && before < S) outp[before] = i;
    cnt += __popc(mask);
    if (cnt >= S) break;        // cnt is wave-uniform (from ballot)
  }
  if (first < 0) first = 0;
  for (int t = cnt + lane; t < S; t += 32) outp[t] = first;
}

// ---------------------------------------------------------------------------
// 3. Gather grouped features into X0 (f16, K-major per position, K padded to 96)
// ---------------------------------------------------------------------------
__global__ void gather_x0(const float* __restrict__ xyz,
                          const float* __restrict__ feature,
                          const float* __restrict__ newxyz,
                          const int* __restrict__ idxb,
                          _Float16* __restrict__ X0, int S) {
  const long pos  = (long)blockIdx.x * blockDim.x + threadIdx.x;
  const long Npos = (long)BB * PP * S;
  if (pos >= Npos) return;
  const int  s  = (int)(pos % S);
  const long bp = pos / S;
  const int  b  = (int)(bp / PP);
  const int  i  = idxb[bp * S + s];
  const float* c  = newxyz + bp * 3;
  const float* pt = xyz + ((long)b * NN + i) * 3;
  _Float16* xr = X0 + pos * 96;
  xr[0] = (_Float16)(pt[0] - c[0]);
  xr[1] = (_Float16)(pt[1] - c[1]);
  xr[2] = (_Float16)(pt[2] - c[2]);
  const float* f = feature + (long)b * CINC * NN + i;
#pragma unroll
  for (int ch = 0; ch < CINC; ++ch) xr[3 + ch] = (_Float16)f[(long)ch * NN];
#pragma unroll
  for (int k = 67; k < 96; ++k) xr[k] = (_Float16)0.0f;
}

// ---------------------------------------------------------------------------
// 4. Weight prep: f32 (Cout x Cin) -> f16 (Cout x KP) zero-padded
// ---------------------------------------------------------------------------
__global__ void wprep(const float* __restrict__ w, _Float16* __restrict__ wh,
                      int Cout, int Cin, int KP) {
  const int t = blockIdx.x * blockDim.x + threadIdx.x;
  if (t >= Cout * KP) return;
  const int k = t % KP, r = t / KP;
  wh[t] = (k < Cin) ? (_Float16)w[r * Cin + k] : (_Float16)0.0f;
}

// ---------------------------------------------------------------------------
// 5. WMMA GEMM: H(Cout x Npos, f16 pos-major) = W(Cout x KP) * X(KP x Npos).
//    4 waves/block, 64 positions/block. W staged in LDS cooperatively; the
//    block's contiguous X tile (64*KP halves) staged into LDS by ONE Tensor
//    Data Mover descriptor (1-D tile, data_size=2B), waited with
//    s_wait_tensorcnt, then both WMMA operands stream from LDS.
// ---------------------------------------------------------------------------
template <int MT>
__global__ __launch_bounds__(128)
void gemm_wmma(const _Float16* __restrict__ W, const _Float16* __restrict__ X,
               _Float16* __restrict__ H, int KP, int Cout, long Npos) {
  extern __shared__ char smem[];
  _Float16* sW = (_Float16*)smem;
  _Float16* sX = sW + (size_t)Cout * KP;
  {
    const int nW8 = Cout * KP / 8;
    const v8h* gW = (const v8h*)W;
    v8h* sW8 = (v8h*)sW;
    for (int i = threadIdx.x; i < nW8; i += blockDim.x) sW8[i] = gW[i];
  }

#if USE_TDM
  {
    // Tensor DMA: contiguous 1-D tile of 64*KP f16 elements -> LDS (sX).
    const unsigned tileElems = (unsigned)(64 * KP);
    const unsigned ldsAddr = (unsigned)(uintptr_t)(void*)sX;
    const unsigned long long ga =
        (unsigned long long)(uintptr_t)(const void*)(X + (long)blockIdx.x * 64 * KP);
    v4u g0;
    g0[0] = 1u;                                            // count=1, user D#
    g0[1] = ldsAddr;                                       // lds_addr (bytes)
    g0[2] = (unsigned)(ga & 0xffffffffu);                  // global_addr[31:0]
    g0[3] = (unsigned)((ga >> 32) & 0x01ffffffu) | (2u << 30); // addr[56:32], type=2
    v8i g1;
    g1[0] = (int)(1u << 16);                               // wg_mask=0, data_size=2B
    g1[1] = (int)((tileElems & 0xffffu) << 16);            // tensor_dim0[15:0]
    g1[2] = (int)(((tileElems >> 16) & 0xffffu) | (1u << 16)); // dim0 hi, tensor_dim1=1
    g1[3] = (int)(tileElems << 16);                        // tile_dim0
    g1[4] = 0;                                             // tile_dim1=0 (1-D), tile_dim2=0
    g1[5] = (int)tileElems;                                // tensor_dim0_stride lo
    g1[6] = 0;                                             // stride hi / dim1_stride lo
    g1[7] = 0;
    v4i g2; g2[0] = 1; g2[1] = 1; g2[2] = 0; g2[3] = 0;    // dims 2/3 = 1, unused
    v4i g3; g3[0] = 0; g3[1] = 0; g3[2] = 0; g3[3] = 0;
    if (threadIdx.x < 32) {                                // one wave issues (EXEC ignored)
#if TDM_ARGS6
      v8i g4;
      g4[0] = 0; g4[1] = 0; g4[2] = 0; g4[3] = 0;
      g4[4] = 0; g4[5] = 0; g4[6] = 0; g4[7] = 0;
      __builtin_amdgcn_tensor_load_to_lds(g0, g1, g2, g3, g4, 0);
#else
      __builtin_amdgcn_tensor_load_to_lds(g0, g1, g2, g3, 0);
#endif
      __builtin_amdgcn_s_wait_tensorcnt(0);
    }
  }
#endif
  __syncthreads();

  const int wave = threadIdx.x >> 5;
  const int lane = threadIdx.x & 31;
  const int nlo = lane & 15;       // A: row M; B: column N; D: column N
  const int kh  = lane >> 4;       // K half selector

  const long n0 = ((long)blockIdx.x * 4 + wave) * 16;

  v8f acc[MT];
#pragma unroll
  for (int mt = 0; mt < MT; ++mt)
#pragma unroll
    for (int i = 0; i < 8; ++i) acc[mt][i] = 0.0f;

#if USE_TDM
  const _Float16* xsrc = sX + (long)(wave * 16 + nlo) * KP;   // B from LDS tile
#else
  const _Float16* xsrc = X + (n0 + nlo) * (long)KP;           // B from global
#endif
  for (int kt = 0; kt < KP; kt += 32) {
#if !USE_TDM
    __builtin_prefetch(xsrc + kt + 64, 0, 1);   // global_prefetch_b8
#endif
    Frag16 bf;                                   // B: lanes 0-15 K=0..15, 16-31 K=16..31
    bf.h[0] = *(const v8h*)(xsrc + kt + kh * 16);
    bf.h[1] = *(const v8h*)(xsrc + kt + kh * 16 + 8);
#pragma unroll
    for (int mt = 0; mt < MT; ++mt) {
      const _Float16* wr = sW + (long)(mt * 16 + nlo) * KP + kt + kh * 16;
      Frag16 af;                                 // A: lanes 0-15 K=0..15, 16-31 K=16..31
      af.h[0] = *(const v8h*)(wr);
      af.h[1] = *(const v8h*)(wr + 8);
      acc[mt] = __builtin_amdgcn_wmma_f32_16x16x32_f16(
          false, af.v, false, bf.v, (short)0, acc[mt], false, false);
    }
  }

  const long n = n0 + nlo;
  const int mbase = kh * 8;                      // D: lanes>=16 hold M+8
#pragma unroll
  for (int mt = 0; mt < MT; ++mt) {
    v8h hv;
#pragma unroll
    for (int i = 0; i < 8; ++i) hv[i] = (_Float16)acc[mt][i];
    *(v8h*)(H + n * Cout + mt * 16 + mbase) = hv;
  }
}

// ---------------------------------------------------------------------------
// 6. BN stats + fold: one block per channel (deterministic tree reduction),
//    emits a,b with out = relu(a*h + b)
// ---------------------------------------------------------------------------
__global__ void stats_bn(const _Float16* __restrict__ H, float* __restrict__ ab,
                         const float* __restrict__ gamma,
                         const float* __restrict__ beta, long Npos, int Cout) {
  const int c = blockIdx.x;
  __shared__ float ss[256], sq[256];
  float s = 0.f, q = 0.f;
  for (long i = threadIdx.x; i < Npos; i += 256) {
    const float v = (float)H[i * Cout + c];
    s += v; q += v * v;
  }
  ss[threadIdx.x] = s; sq[threadIdx.x] = q;
  __syncthreads();
  for (int o = 128; o; o >>= 1) {
    if ((int)threadIdx.x < o) {
      ss[threadIdx.x] += ss[threadIdx.x + o];
      sq[threadIdx.x] += sq[threadIdx.x + o];
    }
    __syncthreads();
  }
  if (threadIdx.x == 0) {
    const float mu  = ss[0] / (float)Npos;
    const float var = sq[0] / (float)Npos - mu * mu;
    const float a   = gamma[c] * rsqrtf(var + EPSB);
    ab[2 * c]     = a;
    ab[2 * c + 1] = beta[c] - mu * a;
  }
}

__global__ void bnrelu(_Float16* __restrict__ H, const float* __restrict__ ab,
                       long total, int Cout) {
  const long e = ((long)blockIdx.x * blockDim.x + threadIdx.x) * 8;
  if (e >= total) return;
  const int k0 = (int)(e % Cout);
#pragma unroll
  for (int j = 0; j < 8; ++j) {
    const float h = (float)H[e + j];
    H[e + j] = (_Float16)fmaxf(0.0f, ab[2 * (k0 + j)] * h + ab[2 * (k0 + j) + 1]);
  }
}

// ---------------------------------------------------------------------------
// 7. Max pool over neighborhood -> concat buffer F (384 channels, pos-major)
// ---------------------------------------------------------------------------
__global__ void maxpool_s(const _Float16* __restrict__ H, _Float16* __restrict__ F,
                          int S, int chOff) {
  const long t = (long)blockIdx.x * blockDim.x + threadIdx.x;
  if (t >= (long)BB * PP * 128) return;
  const int  c  = (int)(t % 128);
  const long bp = t / 128;
  float m = -1e30f;
  const _Float16* hp = H + bp * S * 128 + c;
  for (int s = 0; s < S; ++s) m = fmaxf(m, (float)hp[(long)s * 128]);
  F[bp * 384 + chOff + c] = (_Float16)m;
}

// ---------------------------------------------------------------------------
// 8. Final: BN+ReLU on aggregated features -> d_out (B,128,P) f32
// ---------------------------------------------------------------------------
__global__ void final_out(const _Float16* __restrict__ Y, const float* __restrict__ ab,
                          float* __restrict__ out) {
  const long t = (long)blockIdx.x * blockDim.x + threadIdx.x;
  if (t >= (long)BB * 128 * PP) return;
  const int p = (int)(t % PP);
  const int c = (int)((t / PP) % 128);
  const int b = (int)(t / ((long)PP * 128));
  const float y = (float)Y[((long)b * PP + p) * 128 + c];
  out[t] = fmaxf(0.0f, ab[2 * c] * y + ab[2 * c + 1]);
}

// ---------------------------------------------------------------------------
// Host orchestration
// ---------------------------------------------------------------------------
static void launch_gemm(const _Float16* W, const _Float16* X, _Float16* H,
                        int KP, int Cout, long Npos, hipStream_t st) {
  dim3 g((unsigned)(Npos / 64)), blk(128);
  size_t sh = (size_t)Cout * KP * 2 + (size_t)64 * KP * 2;  // W + TDM X tile
  switch (Cout / 16) {
    case 4:  gemm_wmma<4><<<g, blk, sh, st>>>(W, X, H, KP, Cout, Npos); break;
    case 6:  gemm_wmma<6><<<g, blk, sh, st>>>(W, X, H, KP, Cout, Npos); break;
    default: gemm_wmma<8><<<g, blk, sh, st>>>(W, X, H, KP, Cout, Npos); break;
  }
}

extern "C" void kernel_launch(void* const* d_in, const int* in_sizes, int n_in,
                              void* d_out, int out_size, void* d_ws, size_t ws_size,
                              hipStream_t stream) {
  (void)in_sizes; (void)n_in; (void)out_size; (void)ws_size;
  const float* xyz     = (const float*)d_in[0];
  const float* feature = (const float*)d_in[1];
  // d_in[2] = fps_idx (unused by reference)
  float* out = (float*)d_out;
  float* new_xyz = out;                              // (B,P,3)

  struct LDef { int Cout, Cin, KP; };
  const LDef mlp[3][3] = {
      {{64, 67, 96}, {64, 64, 64}, {128, 64, 64}},
      {{64, 67, 96}, {64, 64, 64}, {128, 64, 64}},
      {{64, 67, 96}, {96, 64, 64}, {128, 96, 96}},
  };
  const float RAD[3] = {0.4f, 0.8f, 1.6f};
  const int   SS[3]  = {16, 32, 64};

  // workspace layout
  char* ws = (char*)d_ws;
  size_t off = 0;
  auto wsalloc = [&](size_t bytes) {
    off = (off + 255) & ~(size_t)255;
    void* p = ws + off;
    off += bytes;
    return p;
  };
  _Float16* Wh[3][3];
  for (int s = 0; s < 3; ++s)
    for (int l = 0; l < 3; ++l)
      Wh[s][l] = (_Float16*)wsalloc((size_t)mlp[s][l].Cout * mlp[s][l].KP * 2);
  _Float16* WhAgg = (_Float16*)wsalloc((size_t)128 * 384 * 2);
  int*      idxb  = (int*)wsalloc((size_t)BB * PP * 64 * 4);
  const size_t bigHalves = (size_t)128 * BB * PP * 64;            // 128 rows x max Npos
  _Float16* PA = (_Float16*)wsalloc(bigHalves * 2);
  _Float16* PB = (_Float16*)wsalloc(bigHalves * 2);
  _Float16* F  = (_Float16*)wsalloc((size_t)BB * PP * 384 * 2);
  float*    ab = (float*)wsalloc(2 * 128 * 4);

  // --- weight conversion/padding ---
  for (int s = 0; s < 3; ++s)
    for (int l = 0; l < 3; ++l) {
      const LDef& L = mlp[s][l];
      const float* w = (const float*)d_in[3 + s * 9 + l * 3 + 0];
      int tot = L.Cout * L.KP;
      wprep<<<(tot + 255) / 256, 256, 0, stream>>>(w, Wh[s][l], L.Cout, L.Cin, L.KP);
    }
  {
    const float* wagg = (const float*)d_in[30];
    wprep<<<(128 * 384 + 255) / 256, 256, 0, stream>>>(wagg, WhAgg, 128, 384, 384);
  }

  // --- FPS (writes new_xyz and inds regions of d_out) ---
  fps_kernel<<<BB, 1024, 3 * NN * sizeof(float), stream>>>(xyz, out);

  // --- per-scale: ball query -> gather -> 3x (GEMM, stats, BN+ReLU) -> maxpool
  for (int s = 0; s < 3; ++s) {
    const int  S    = SS[s];
    const long Npos = (long)BB * PP * S;
    ball_query<<<(BB * PP) / 8, 256, 0, stream>>>(xyz, new_xyz, idxb,
                                                  RAD[s] * RAD[s], S);
    gather_x0<<<(unsigned)((Npos + 255) / 256), 256, 0, stream>>>(
        xyz, feature, new_xyz, idxb, PA, S);

    _Float16* Xc = PA;
    _Float16* Hc = PB;
    for (int l = 0; l < 3; ++l) {
      const LDef& L = mlp[s][l];
      const float* gamma = (const float*)d_in[3 + s * 9 + l * 3 + 1];
      const float* beta  = (const float*)d_in[3 + s * 9 + l * 3 + 2];
      launch_gemm(Wh[s][l], Xc, Hc, L.KP, L.Cout, Npos, stream);
      stats_bn<<<L.Cout, 256, 0, stream>>>(Hc, ab, gamma, beta, Npos, L.Cout);
      const long tot = Npos * L.Cout;
      bnrelu<<<(unsigned)(tot / 8 / 256), 256, 0, stream>>>(Hc, ab, tot, L.Cout);
      _Float16* t = Xc; Xc = Hc; Hc = t;       // ping-pong
    }
    // final layer output (post BN+ReLU) is in Xc after the last swap
    maxpool_s<<<(unsigned)(((long)BB * PP * 128 + 255) / 256), 256, 0, stream>>>(
        Xc, F, S, s * 128);
  }

  // --- aggregation conv1d (bias cancels inside BN) + BN + ReLU -> d_out ---
  _Float16* Yagg = PA;
  launch_gemm(WhAgg, F, Yagg, 384, 128, (long)BB * PP, stream);
  stats_bn<<<128, 256, 0, stream>>>(Yagg, ab, (const float*)d_in[32],
                                    (const float*)d_in[33], (long)BB * PP, 128);
  final_out<<<(unsigned)(((long)BB * 128 * PP + 255) / 256), 256, 0, stream>>>(
      Yagg, ab, out + (size_t)BB * PP * 3);
}
